// transformer_snac_block_5488968204587
// MI455X (gfx1250) — compile-verified, matrix operand order checked by reading, not run on hardware
//
#include <hip/hip_runtime.h>
#include <hip/hip_bf16.h>

// ---------------------------------------------------------------------------
// Types for CDNA5 WMMA (wave32): V_WMMA_F32_16X16X32_BF16
// ---------------------------------------------------------------------------
typedef __attribute__((ext_vector_type(16))) __bf16    v16bf;
typedef __attribute__((ext_vector_type(8)))  float     v8f;
typedef __attribute__((ext_vector_type(8)))  unsigned  v8u;

#define BB 4
#define TT 1024
#define DD 1024
#define HH 16
#define HDD 64
#define TPP 1024
#define NSS 2048          // TP + T
#define BT (BB * TT)      // 4096

static __device__ __forceinline__ __bf16 f2bf(float x) {
    union { float f; unsigned u; } c; c.f = x;
    unsigned r = (c.u + 0x7FFFu + ((c.u >> 16) & 1u)) >> 16;
    return __builtin_bit_cast(__bf16, (unsigned short)r);
}

// Async global -> LDS copy of 16 bytes per lane (CDNA5 ASYNCcnt path).
// CDNA5 ISA: the instruction offset is added to both the LDS and the global address
static __device__ __forceinline__ void async_b128(unsigned lds_off, const void* gptr) {
    asm volatile("global_load_async_to_lds_b128 %0, %1, off"
                 :: "v"(lds_off), "v"((unsigned long long)(uintptr_t)gptr)
                 : "memory");
}
static __device__ __forceinline__ void async_b128_off16(unsigned lds_off, const void* gptr) {
    asm volatile("global_load_async_to_lds_b128 %0, %1, off offset:16"
                 :: "v"(lds_off), "v"((unsigned long long)(uintptr_t)gptr)
                 : "memory");
}
static __device__ __forceinline__ void wait_async0() {
    asm volatile("s_wait_asynccnt 0x0" ::: "memory");
}

// Low 32 bits of a generic pointer to __shared__ = LDS byte offset (aperture rule)
static __device__ __forceinline__ unsigned lds_off_of(const void* p) {
    return (unsigned)(uintptr_t)p;
}

// Fragment load (A-layout and B^T-rows layout are identical for bf16 16x16x32):
// lane L holds row (L&15); K pattern: half=(L>>4): dwords {0,1,2,3} -> K 0..7 (+8*half),
// dwords {4..7} -> K 16..23 (+8*half).
static __device__ __forceinline__ v16bf frag_ld(const __bf16* base, int stride, int lane) {
    const int r = lane & 15, h = lane >> 4;
    const unsigned* p = (const unsigned*)(base + r * stride + h * 8);
    v8u d;
#pragma unroll
    for (int g = 0; g < 8; ++g) {
        const int di = (g < 4) ? g : (8 + (g - 4));  // dwords {0,1,2,3, 8,9,10,11}
        d[g] = p[di];
    }
    return __builtin_bit_cast(v16bf, d);
}

struct Acc22 { v8f a[2][2]; };

static __device__ __forceinline__ void acc_zero(Acc22& A) {
#pragma unroll
    for (int i = 0; i < 2; ++i)
#pragma unroll
        for (int j = 0; j < 2; ++j) A.a[i][j] = (v8f){0.f,0.f,0.f,0.f,0.f,0.f,0.f,0.f};
}

// ---------------------------------------------------------------------------
// Shared main loop: block tile 64(M) x 128(N), 8 waves (2 in M x 4 in N),
// each wave computes a 32x32 tile = 2x2 WMMA 16x16x32_bf16 tiles.
// A: row-major (M x K), Bt: row-major (N x K)  [i.e. B stored transposed].
// DOUBLE-BUFFERED async pipeline: tile k+1 streams global->LDS via
// GLOBAL_LOAD_ASYNC_TO_LDS_B128 while the WMMAs consume tile k.
// One barrier per K-step; ASYNCcnt in-order completion makes the top-of-loop
// s_wait_asynccnt 0 a fully-overlapped wait.
// LDS stride 40 elems (80 B -> rows stay 16 B aligned).
// ---------------------------------------------------------------------------
#define LDSTR 40
#define ATILE (64 * LDSTR)
#define BTILE (128 * LDSTR)

static __device__ __forceinline__ void gemm_mainloop_64x128(
    const __bf16* __restrict__ Ag, int lda,
    const __bf16* __restrict__ Bg, int ldb,
    int K, int m0, int n0,
    __bf16* ldsA /*2 x ATILE*/, __bf16* ldsB /*2 x BTILE*/, Acc22& acc)
{
    const int t    = threadIdx.x;
    const int lane = t & 31;
    const int w    = t >> 5;
    const int wm   = w & 1;      // 0..1  (M)
    const int wn   = w >> 1;     // 0..3  (N)

    // per-thread staging coordinates (invariant over K loop)
    const int ar  = t >> 2, acg = (t & 3) << 3;          // A: 8 elems (16 B)
    const int br  = t >> 1, bcg = (t & 1) << 4;          // B: 16 elems (2 x 16 B)
    const __bf16* aAddr = Ag + (size_t)(m0 + ar) * lda + acg;
    const __bf16* bAddr = Bg + (size_t)(n0 + br) * ldb + bcg;
    const unsigned aoff[2] = { lds_off_of(ldsA + ar * LDSTR + acg),
                               lds_off_of(ldsA + ATILE + ar * LDSTR + acg) };
    const unsigned boff[2] = { lds_off_of(ldsB + br * LDSTR + bcg),
                               lds_off_of(ldsB + BTILE + br * LDSTR + bcg) };

    // prologue: tile 0 -> buffer 0
    async_b128(aoff[0], aAddr);
    async_b128(boff[0], bAddr);
    async_b128_off16(boff[0], bAddr);

    int kb = 0;
    for (int k0 = 0; k0 < K; k0 += 32, kb ^= 1) {
        wait_async0();        // tile k landed (per-wave, in-order)
        __syncthreads();      // all waves see tile k; tile k-1 reads all drained
        if (k0 + 32 < K) {    // stream tile k+1 while computing tile k
            async_b128(aoff[kb ^ 1], aAddr + k0 + 32);
            const __bf16* bs = bAddr + k0 + 32;
            async_b128(boff[kb ^ 1], bs);
            async_b128_off16(boff[kb ^ 1], bs);
        }

        const __bf16* bufA = ldsA + kb * ATILE;
        const __bf16* bufB = ldsB + kb * BTILE;
        v16bf af[2], bfv[2];
#pragma unroll
        for (int mi = 0; mi < 2; ++mi)
            af[mi] = frag_ld(bufA + (wm * 32 + mi * 16) * LDSTR, LDSTR, lane);
#pragma unroll
        for (int ni = 0; ni < 2; ++ni)
            bfv[ni] = frag_ld(bufB + (wn * 32 + ni * 16) * LDSTR, LDSTR, lane);

#pragma unroll
        for (int mi = 0; mi < 2; ++mi)
#pragma unroll
            for (int ni = 0; ni < 2; ++ni)
                acc.a[mi][ni] = __builtin_amdgcn_wmma_f32_16x16x32_bf16(
                    false, af[mi], false, bfv[ni], (short)0, acc.a[mi][ni], false, false);
    }
}

// C/D 16x16 f32 layout: element v of frag -> m = 8*(lane>>4)+v, n = lane&15.
#define EPI_COORDS(mi, ni, v, m, n)                                   \
    const int m = wm * 32 + (mi) * 16 + ((lane >> 4) << 3) + (v);     \
    const int n = wn * 32 + (ni) * 16 + (lane & 15);

// ---------------------------------------------------------------------------
// Kernel: weight prep  w = tanh(wt)*sigmoid(mt), stored TRANSPOSED bf16
// ---------------------------------------------------------------------------
__global__ __launch_bounds__(256) void prep_weight_kernel(
    const float* __restrict__ wt, const float* __restrict__ mt,
    __bf16* __restrict__ outT, int cols)
{
    const int idx = blockIdx.x * 256 + threadIdx.x;
    const int total = DD * cols;
    if (idx >= total) return;
    const int r = idx / cols, c = idx - r * cols;
    const float w = tanhf(wt[idx]) * (1.0f / (1.0f + __expf(-mt[idx])));
    outT[(size_t)c * DD + r] = f2bf(w);
}

// ---------------------------------------------------------------------------
// Kernel: layernorm over D=1024, one block per row, bf16 output
// ---------------------------------------------------------------------------
__global__ __launch_bounds__(256) void layernorm_bf16_kernel(
    const float* __restrict__ x, const float* __restrict__ gamma,
    const float* __restrict__ beta, __bf16* __restrict__ out)
{
    __shared__ float red[256];
    const int row = blockIdx.x;
    const float* xr = x + (size_t)row * DD;
    float v[4]; float s = 0.f;
#pragma unroll
    for (int i = 0; i < 4; ++i) { v[i] = xr[threadIdx.x + i * 256]; s += v[i]; }
    red[threadIdx.x] = s; __syncthreads();
    for (int o = 128; o > 0; o >>= 1) {
        if ((int)threadIdx.x < o) red[threadIdx.x] += red[threadIdx.x + o];
        __syncthreads();
    }
    const float mu = red[0] * (1.f / DD);
    __syncthreads();
    float ss = 0.f;
#pragma unroll
    for (int i = 0; i < 4; ++i) { const float d = v[i] - mu; ss += d * d; }
    red[threadIdx.x] = ss; __syncthreads();
    for (int o = 128; o > 0; o >>= 1) {
        if ((int)threadIdx.x < o) red[threadIdx.x] += red[threadIdx.x + o];
        __syncthreads();
    }
    const float inv = rsqrtf(red[0] * (1.f / DD) + 1e-3f);
#pragma unroll
    for (int i = 0; i < 4; ++i) {
        const int c = threadIdx.x + i * 256;
        out[(size_t)row * DD + c] = f2bf((v[i] - mu) * inv * gamma[c] + beta[c]);
    }
}

// ---------------------------------------------------------------------------
// Kernel: convert past KV (B,2,H,TP,HD) f32 -> bf16 into kfull/vfull rows 0..TP
// ---------------------------------------------------------------------------
__global__ __launch_bounds__(256) void past_cvt_kernel(
    const float* __restrict__ past, __bf16* __restrict__ kfull, __bf16* __restrict__ vfull)
{
    const size_t idx = (size_t)blockIdx.x * 256 + threadIdx.x;
    const int d  = idx & 63;
    const int p  = (idx >> 6) & 1023;
    const int h  = (idx >> 16) & 15;
    const int kv = (idx >> 20) & 1;
    const int b  = (int)(idx >> 21);
    const size_t dst = (((size_t)(b * HH + h) * NSS) + p) * HDD + d;
    const __bf16 val = f2bf(past[idx]);
    if (kv == 0) kfull[dst] = val; else vfull[dst] = val;
}

// ---------------------------------------------------------------------------
// Kernel: QKV GEMM  (4096 x 1024) @ (1024 x 3072)  -> scatter q/k/v + present
// ---------------------------------------------------------------------------
__global__ __launch_bounds__(256) void gemm_qkv_kernel(
    const __bf16* __restrict__ xn, const __bf16* __restrict__ wT,
    __bf16* __restrict__ qbuf, __bf16* __restrict__ kfull, __bf16* __restrict__ vfull,
    float* __restrict__ present)
{
    __shared__ alignas(16) __bf16 ldsA[2 * ATILE];
    __shared__ alignas(16) __bf16 ldsB[2 * BTILE];
    const int m0 = blockIdx.y * 64, n0 = blockIdx.x * 128;
    Acc22 acc; acc_zero(acc);
    gemm_mainloop_64x128(xn, DD, wT, DD, DD, m0, n0, ldsA, ldsB, acc);

    const int lane = threadIdx.x & 31, w = threadIdx.x >> 5, wm = w & 1, wn = w >> 1;
#pragma unroll
    for (int mi = 0; mi < 2; ++mi)
#pragma unroll
        for (int ni = 0; ni < 2; ++ni)
#pragma unroll
            for (int v = 0; v < 8; ++v) {
                EPI_COORDS(mi, ni, v, ml, nl);
                const int gm = m0 + ml, gn = n0 + nl;
                const float val = acc.a[mi][ni][v];
                const int b = gm >> 10, tq = gm & 1023;
                const int sec = gn >> 10, hn = gn & 1023;
                const int h = hn >> 6, d = hn & 63;
                if (sec == 0) {
                    qbuf[(((size_t)(b * HH + h) * TT + tq) << 6) + d] = f2bf(val);
                } else {
                    const size_t kvi = (((size_t)(b * HH + h) * NSS) + TPP + tq) * HDD + d;
                    if (sec == 1) kfull[kvi] = f2bf(val); else vfull[kvi] = f2bf(val);
                    const size_t pi = ((((size_t)(b * 2 + (sec - 1)) * HH + h) * TT + tq) << 6) + d;
                    present[pi] = val;
                }
            }
}

// ---------------------------------------------------------------------------
// Kernel: scores GEMM per (b,h): q(1024x64) @ k^T(64x2048), scale + causal mask
// ---------------------------------------------------------------------------
__global__ __launch_bounds__(256) void gemm_scores_kernel(
    const __bf16* __restrict__ qbuf, const __bf16* __restrict__ kfull,
    float* __restrict__ msk)
{
    __shared__ alignas(16) __bf16 ldsA[2 * ATILE];
    __shared__ alignas(16) __bf16 ldsB[2 * BTILE];
    const int bh = blockIdx.z;
    const __bf16* A = qbuf  + (size_t)bh * TT * HDD;
    const __bf16* B = kfull + (size_t)bh * NSS * HDD;
    const int m0 = blockIdx.y * 64, n0 = blockIdx.x * 128;
    Acc22 acc; acc_zero(acc);
    gemm_mainloop_64x128(A, HDD, B, HDD, HDD, m0, n0, ldsA, ldsB, acc);

    const int lane = threadIdx.x & 31, w = threadIdx.x >> 5, wm = w & 1, wn = w >> 1;
    float* row0 = msk + (size_t)bh * TT * NSS;
#pragma unroll
    for (int mi = 0; mi < 2; ++mi)
#pragma unroll
        for (int ni = 0; ni < 2; ++ni)
#pragma unroll
            for (int v = 0; v < 8; ++v) {
                EPI_COORDS(mi, ni, v, ml, nl);
                const int i = m0 + ml, j = n0 + nl;
                const float s = acc.a[mi][ni][v];
                row0[(size_t)i * NSS + j] = (j <= i + TPP) ? s * 0.125f : -1e10f;
            }
}

// ---------------------------------------------------------------------------
// Kernel: in-place softmax over rows of 2048 (block per row)
// ---------------------------------------------------------------------------
__global__ __launch_bounds__(256) void softmax_kernel(float* __restrict__ msk)
{
    __shared__ float red[256];
    float* row = msk + (size_t)blockIdx.x * NSS;
    float v[8]; float mx = -3.4e38f;
#pragma unroll
    for (int i = 0; i < 8; ++i) { v[i] = row[threadIdx.x + i * 256]; mx = fmaxf(mx, v[i]); }
    red[threadIdx.x] = mx; __syncthreads();
    for (int o = 128; o > 0; o >>= 1) {
        if ((int)threadIdx.x < o) red[threadIdx.x] = fmaxf(red[threadIdx.x], red[threadIdx.x + o]);
        __syncthreads();
    }
    mx = red[0]; __syncthreads();
    float s = 0.f;
#pragma unroll
    for (int i = 0; i < 8; ++i) { v[i] = __expf(v[i] - mx); s += v[i]; }
    red[threadIdx.x] = s; __syncthreads();
    for (int o = 128; o > 0; o >>= 1) {
        if ((int)threadIdx.x < o) red[threadIdx.x] += red[threadIdx.x + o];
        __syncthreads();
    }
    const float inv = 1.0f / red[0];
#pragma unroll
    for (int i = 0; i < 8; ++i) row[threadIdx.x + i * 256] = v[i] * inv;
}

// ---------------------------------------------------------------------------
// Kernel: PV GEMM per (b,h): P(1024x2048 f32 -> bf16 on the fly) @ V(2048x64)
// block tile 128(M) x 64(N), 8 waves = 4(M) x 2(N), wave tile 32x32.
// V staged TRANSPOSED into LDS (manual; conversion/transpose can't use async).
// ---------------------------------------------------------------------------
__global__ __launch_bounds__(256) void gemm_av_kernel(
    const float* __restrict__ msk, const __bf16* __restrict__ vfull,
    __bf16* __restrict__ ah)
{
    __shared__ alignas(16) __bf16 ldsA[128 * LDSTR];
    __shared__ alignas(16) __bf16 ldsB[64 * LDSTR];
    const int bh = blockIdx.y;
    const int m0 = blockIdx.x * 128;
    const float* P = msk + (size_t)bh * TT * NSS;
    const __bf16* V = vfull + (size_t)bh * NSS * HDD;

    const int t = threadIdx.x, lane = t & 31, w = t >> 5;
    const int wm = w & 3, wn = w >> 2;
    Acc22 acc; acc_zero(acc);

    for (int k0 = 0; k0 < NSS; k0 += 32) {
        {   // stage A: 128 x 32, f32 -> bf16 (16 floats per thread)
            const int r = t >> 1, cg = (t & 1) << 4;
            const float* src = P + (size_t)(m0 + r) * NSS + k0 + cg;
            __bf16* dst = ldsA + r * LDSTR + cg;
#pragma unroll
            for (int q = 0; q < 16; q += 4) {
                float4 f = *(const float4*)(src + q);
                dst[q + 0] = f2bf(f.x); dst[q + 1] = f2bf(f.y);
                dst[q + 2] = f2bf(f.z); dst[q + 3] = f2bf(f.w);
            }
        }
        {   // stage B transposed: ldsB[n][k] = V[k0+k][n]   (8 elems per thread)
            const int k = t & 31, ng = (t >> 5) << 3;
            uint4 raw = *(const uint4*)(V + (size_t)(k0 + k) * HDD + ng);
            const __bf16* e = (const __bf16*)&raw;
#pragma unroll
            for (int j = 0; j < 8; ++j) ldsB[(ng + j) * LDSTR + k] = e[j];
        }
        __syncthreads();

        v16bf af[2], bfv[2];
#pragma unroll
        for (int mi = 0; mi < 2; ++mi)
            af[mi] = frag_ld(ldsA + (wm * 32 + mi * 16) * LDSTR, LDSTR, lane);
#pragma unroll
        for (int ni = 0; ni < 2; ++ni)
            bfv[ni] = frag_ld(ldsB + (wn * 32 + ni * 16) * LDSTR, LDSTR, lane);
#pragma unroll
        for (int mi = 0; mi < 2; ++mi)
#pragma unroll
            for (int ni = 0; ni < 2; ++ni)
                acc.a[mi][ni] = __builtin_amdgcn_wmma_f32_16x16x32_bf16(
                    false, af[mi], false, bfv[ni], (short)0, acc.a[mi][ni], false, false);
        __syncthreads();
    }

    const int b = bh >> 4, h = bh & 15;
#pragma unroll
    for (int mi = 0; mi < 2; ++mi)
#pragma unroll
        for (int ni = 0; ni < 2; ++ni)
#pragma unroll
            for (int v = 0; v < 8; ++v) {
                const int ml = wm * 32 + mi * 16 + ((lane >> 4) << 3) + v;
                const int nl = wn * 32 + ni * 16 + (lane & 15);   // 0..63 = head dim
                const size_t bt = (size_t)b * TT + (m0 + ml);
                ah[bt * DD + h * HDD + nl] = f2bf(acc.a[mi][ni][v]);
            }
}

// ---------------------------------------------------------------------------
// Kernel: GEMM (4096x1024)@(1024x1024) + elementwise residual add
// ---------------------------------------------------------------------------
__global__ __launch_bounds__(256) void gemm_add_kernel(
    const __bf16* __restrict__ A, const __bf16* __restrict__ Bt,
    const float* __restrict__ addend, float* __restrict__ out)
{
    __shared__ alignas(16) __bf16 ldsA[2 * ATILE];
    __shared__ alignas(16) __bf16 ldsB[2 * BTILE];
    const int m0 = blockIdx.y * 64, n0 = blockIdx.x * 128;
    Acc22 acc; acc_zero(acc);
    gemm_mainloop_64x128(A, DD, Bt, DD, DD, m0, n0, ldsA, ldsB, acc);

    const int lane = threadIdx.x & 31, w = threadIdx.x >> 5, wm = w & 1, wn = w >> 1;
#pragma unroll
    for (int mi = 0; mi < 2; ++mi)
#pragma unroll
        for (int ni = 0; ni < 2; ++ni)
#pragma unroll
            for (int v = 0; v < 8; ++v) {
                EPI_COORDS(mi, ni, v, ml, nl);
                const size_t idx = (size_t)(m0 + ml) * DD + (n0 + nl);
                out[idx] = addend[idx] + acc.a[mi][ni][v];
            }
}

// ---------------------------------------------------------------------------
// Host launcher
// ---------------------------------------------------------------------------
extern "C" void kernel_launch(void* const* d_in, const int* in_sizes, int n_in,
                              void* d_out, int out_size, void* d_ws, size_t ws_size,
                              hipStream_t stream) {
    (void)in_sizes; (void)n_in; (void)out_size; (void)ws_size;
    const float* input    = (const float*)d_in[0];
    const float* past     = (const float*)d_in[1];
    const float* qkv_wt   = (const float*)d_in[2];
    const float* qkv_mt   = (const float*)d_in[3];
    const float* merge_wt = (const float*)d_in[4];
    const float* merge_mt = (const float*)d_in[5];
    const float* ln1_wt   = (const float*)d_in[6];
    const float* ln1_mt   = (const float*)d_in[7];
    const float* g1       = (const float*)d_in[8];
    const float* b1       = (const float*)d_in[9];
    const float* g2       = (const float*)d_in[10];
    const float* b2       = (const float*)d_in[11];

    float* out = (float*)d_out;
    const size_t PRESENT_OFF = (size_t)BB * TT * DD;                            // 4M
    const size_t MSK_OFF     = PRESENT_OFF + (size_t)BB * 2 * HH * TT * HDD;    // 12.5M
    float* out_xm      = out;
    float* out_present = out + PRESENT_OFF;
    float* out_msk     = out + MSK_OFF;

    // Workspace carve-out
    char* p = (char*)d_ws;
    auto carve = [&](size_t bytes) -> void* {
        void* r = (void*)p; p += (bytes + 255) & ~(size_t)255; return r;
    };
    __bf16* wqkvT   = (__bf16*)carve((size_t)3 * DD * DD * 2);   // (3072 x 1024)
    __bf16* wmergeT = (__bf16*)carve((size_t)DD * DD * 2);
    __bf16* wlnT    = (__bf16*)carve((size_t)DD * DD * 2);
    __bf16* xn      = (__bf16*)carve((size_t)BT * DD * 2);
    __bf16* qbuf    = (__bf16*)carve((size_t)BB * HH * TT * HDD * 2);
    __bf16* kfull   = (__bf16*)carve((size_t)BB * HH * NSS * HDD * 2);
    __bf16* vfull   = (__bf16*)carve((size_t)BB * HH * NSS * HDD * 2);
    __bf16* ahbuf   = (__bf16*)carve((size_t)BT * DD * 2);
    float*  xa      = (float*) carve((size_t)BT * DD * 4);
    __bf16* xaln    = (__bf16*)carve((size_t)BT * DD * 2);

    // 1) SNAC weight prep (transposed bf16)
    prep_weight_kernel<<<(DD * 3 * DD + 255) / 256, 256, 0, stream>>>(qkv_wt, qkv_mt, wqkvT, 3 * DD);
    prep_weight_kernel<<<(DD * DD + 255) / 256, 256, 0, stream>>>(merge_wt, merge_mt, wmergeT, DD);
    prep_weight_kernel<<<(DD * DD + 255) / 256, 256, 0, stream>>>(ln1_wt, ln1_mt, wlnT, DD);

    // 2) layernorm(input) -> bf16
    layernorm_bf16_kernel<<<BT, 256, 0, stream>>>(input, g1, b1, xn);

    // 3) past KV -> bf16 rows [0, TP)
    past_cvt_kernel<<<(BB * 2 * HH * TPP * HDD) / 256, 256, 0, stream>>>(past, kfull, vfull);

    // 4) QKV GEMM: scatter q / k / v, write `present`
    gemm_qkv_kernel<<<dim3(3 * DD / 128, BT / 64), 256, 0, stream>>>(
        xn, wqkvT, qbuf, kfull, vfull, out_present);

    // 5) scores GEMM (scaled + causal mask) -> msk region (raw logits)
    gemm_scores_kernel<<<dim3(NSS / 128, TT / 64, BB * HH), 256, 0, stream>>>(
        qbuf, kfull, out_msk);

    // 6) softmax in place
    softmax_kernel<<<BB * HH * TT, 256, 0, stream>>>(out_msk);

    // 7) PV GEMM -> merged-head activations (bf16)
    gemm_av_kernel<<<dim3(TT / 128, BB * HH), 256, 0, stream>>>(out_msk, vfull, ahbuf);

    // 8) merge GEMM + residual: xa = input + ah @ Wmerge
    gemm_add_kernel<<<dim3(DD / 128, BT / 64), 256, 0, stream>>>(ahbuf, wmergeT, input, xa);

    // 9) layernorm(xa) -> bf16
    layernorm_bf16_kernel<<<BT, 256, 0, stream>>>(xa, g2, b2, xaln);

    // 10) MLP GEMM + residual: xm = xa + ln(xa) @ Wln1  -> final output
    gemm_add_kernel<<<dim3(DD / 128, BT / 64), 256, 0, stream>>>(xaln, wlnT, xa, out_xm);
}
